// Attention_17557826306750
// MI455X (gfx1250) — matrix operand order repaired and checked by reading, not compiled, not run on hardware
//
#include <hip/hip_runtime.h>
#include <hip/hip_bf16.h>
#include <stdint.h>

typedef _Float16 v16h __attribute__((ext_vector_type(16)));
typedef float    v8f  __attribute__((ext_vector_type(8)));
typedef int      v4i_ __attribute__((ext_vector_type(4)));

#define C_     192
#define NHEADS 4
#define HD     48
#define HDP    64
#define HH     56
#define NPIX   3136   // 56*56
#define MPIX   784    // 28*28
#define MPAD   800    // 25 * 32
#define BATCH  8

// -------------------------------------------------- async global->LDS staging

#if defined(__has_builtin)
#if __has_builtin(__builtin_amdgcn_global_load_async_to_lds_b128)
#define HAVE_ASYNC_LDS 1
#endif
#endif
#ifndef HAVE_ASYNC_LDS
#define HAVE_ASYNC_LDS 0
#endif

// copy 16 bytes (8 halves) global -> LDS, per-lane
__device__ __forceinline__ void stage16(const _Float16* g, _Float16* l) {
#if HAVE_ASYNC_LDS
  __builtin_amdgcn_global_load_async_to_lds_b128((v4i_*)g, (v4i_*)l, 0, 0);
#else
  *(uint4*)l = *(const uint4*)g;
#endif
}

__device__ __forceinline__ void stage_join() {
#if HAVE_ASYNC_LDS
#if __has_builtin(__builtin_amdgcn_s_wait_asynccnt)
  __builtin_amdgcn_s_wait_asynccnt(0);
#else
  asm volatile("s_wait_asynccnt 0x0" ::: "memory");
#endif
#endif
}

// ---------------------------------------------------------------- WMMA helpers

__device__ __forceinline__ v8f wmma32f16(v16h a, v16h b, v8f c) {
  // D = A(16x32) * B(32x16) + C, f32 accumulate
  return __builtin_amdgcn_wmma_f32_16x16x32_f16(false, a, false, b, (short)0, c,
                                                false, false);
}

// A-fragment (16x32 f16) per ISA 7.12.2: lane row = l&15;
// lanes<16 hold K {0..7,16..23}, lanes>=16 hold K {8..15,24..31}
__device__ __forceinline__ v16h lda(const _Float16* row, int hi) {
  v16h a;
#pragma unroll
  for (int j = 0; j < 16; ++j) {
    int p  = j >> 1;
    int kk = ((p < 4) ? 0 : 16) + hi * 8 + (p & 3) * 2 + (j & 1);
    a[j] = row[kk];
  }
  return a;
}

// B-fragment (32x16 f16): lane col = l&15; lanes<16 hold K {0..7,16..23},
// lanes>=16 hold K {8..15,24..31} (same order as lda).
__device__ __forceinline__ v16h ldb(const _Float16* p, int hi) {
  union { uint4 u[2]; v16h v; } t;
  const uint4* q = (const uint4*)(p + hi * 8);
  t.u[0] = q[0];
  t.u[1] = q[2];
  return t.v;
}

// ---------------------------------------------------------------- prep kernels

__global__ void fold_bn_kernel(const float* g1, const float* b1, const float* m1,
                               const float* v1, const float* g2, const float* b2,
                               const float* m2, const float* v2, float* sc1,
                               float* bi1, float* sc2, float* bi2) {
  int c = threadIdx.x;  // 192
  float s = g1[c] * rsqrtf(v1[c] + 1e-5f);
  sc1[c] = s;
  bi1[c] = b1[c] - m1[c] * s;
  s = g2[c] * rsqrtf(v2[c] + 1e-5f);
  sc2[c] = s;
  bi2[c] = b2[c] - m2[c] * s;
}

// W_eff[o][i] = sum_j W[o][j] * pw[j][i]  (pointwise conv folded into projection)
__global__ void fold_w_kernel(const float* Wq, const float* Wk, const float* Wv,
                              const float* pwq, const float* pwkv, _Float16* WqE,
                              _Float16* WkE, _Float16* WvE) {
  int i = threadIdx.x;   // 192
  int o = blockIdx.x;    // 192
  int which = blockIdx.y;
  float acc = 0.f;
  if (which == 0) {
    for (int j = 0; j < C_; ++j) acc += Wq[o * C_ + j] * pwq[j * C_ + i];
    WqE[o * C_ + i] = (_Float16)acc;
  } else if (which == 1) {
    for (int j = 0; j < C_; ++j) acc += Wk[o * C_ + j] * pwkv[j * C_ + i];
    WkE[o * C_ + i] = (_Float16)acc;
  } else {
    for (int j = 0; j < C_; ++j) acc += Wv[o * C_ + j] * pwkv[(C_ + j) * C_ + i];
    WvE[o * C_ + i] = (_Float16)acc;
  }
}

// --------------------------------------------------- depthwise 3x3 + BN (f16 out)

__global__ __launch_bounds__(256) void dwconv_bn_kernel(
    const float* __restrict__ x, const float* __restrict__ dw,
    const float* __restrict__ bnsc, const float* __restrict__ bnbi,
    _Float16* __restrict__ z, int stride, int outHW) {
  int g = blockIdx.x * blockDim.x + threadIdx.x;
  int c4 = g % (C_ / 4);
  int rest = g / (C_ / 4);
  int outN = outHW * outHW;
  int n = rest % outN;
  int b = rest / outN;
  if (b >= BATCH) return;
  int yo = n / outHW, xo = n % outHW;
  int cb = c4 * 4;
  float acc[4] = {0.f, 0.f, 0.f, 0.f};
#pragma unroll
  for (int dy = 0; dy < 3; ++dy) {
    int yy = yo * stride + dy - 1;
    if ((unsigned)yy >= (unsigned)HH) continue;
#pragma unroll
    for (int dx = 0; dx < 3; ++dx) {
      int xx = xo * stride + dx - 1;
      if ((unsigned)xx >= (unsigned)HH) continue;
      const float4 v =
          *(const float4*)(x + ((size_t)b * NPIX + yy * HH + xx) * C_ + cb);
      const float* vv = (const float*)&v;
#pragma unroll
      for (int ch = 0; ch < 4; ++ch)
        acc[ch] += vv[ch] * dw[(cb + ch) * 9 + dy * 3 + dx];
    }
  }
#pragma unroll
  for (int ch = 0; ch < 4; ++ch) {
    float r = acc[ch] * bnsc[cb + ch] + bnbi[cb + ch];
    z[((size_t)b * outN + n) * C_ + cb + ch] = (_Float16)r;
  }
}

// --------------------------------------------- WMMA GEMM: out = z @ W_eff^T + bias
// rows = B*rowsPerB, cols = 192; output scattered into [b][h][n][64] f16 layout,
// or transposed [b][h][d][mPad] when vT != 0 (for the attention V operand).

__global__ __launch_bounds__(128) void gemm_qkv_kernel(
    const _Float16* __restrict__ z, const _Float16* __restrict__ wt,
    const float* __restrict__ bias, _Float16* __restrict__ dst, int rowsPerB,
    int mPad, int vT) {
  __shared__ _Float16 lA[64 * 32];
  __shared__ _Float16 lB[64 * 32];
  const int rowBase = blockIdx.x * 64;
  const int colBase = blockIdx.y * 64;
  const int tid = threadIdx.x;
  const int w = tid >> 5, l = tid & 31;
  const int lane16 = l & 15, hi = l >> 4;

  v8f acc[4] = {{}, {}, {}, {}};
#pragma unroll 1
  for (int ks = 0; ks < 6; ++ks) {
    for (int i = tid; i < 256; i += 128) {  // 64 rows * 32 halves = 256 x 16B
      int r = i >> 2, p4 = i & 3;
      stage16(z + (size_t)(rowBase + r) * C_ + ks * 32 + p4 * 8, &lA[i * 8]);
      stage16(wt + (size_t)(colBase + r) * C_ + ks * 32 + p4 * 8, &lB[i * 8]);
    }
    stage_join();
    __syncthreads();
    v16h a = lda(&lA[(w * 16 + lane16) * 32], hi);
#pragma unroll
    for (int t = 0; t < 4; ++t) {
      v16h bf = ldb(&lB[(t * 16 + lane16) * 32], hi);
      acc[t] = wmma32f16(a, bf, acc[t]);
    }
    __syncthreads();
  }
#pragma unroll
  for (int t = 0; t < 4; ++t) {
    int o = colBase + t * 16 + lane16;
    float bs = bias[o];
    int h = o / HD, d = o - h * HD;
#pragma unroll
    for (int e = 0; e < 8; ++e) {
      int r = rowBase + w * 16 + e + hi * 8;
      int b_ = r / rowsPerB;
      int n = r - b_ * rowsPerB;
      size_t idx =
          vT ? (((size_t)b_ * NHEADS + h) * HDP + d) * mPad + n
             : (((size_t)b_ * NHEADS + h) * mPad + n) * HDP + d;
      dst[idx] = (_Float16)(acc[t][e] + bs);
    }
  }
}

// ------------------------------------------------------- flash attention kernel
// block = 128 threads (4 waves); wave handles 16 q rows; K/V tiles of 32 shared.

__global__ __launch_bounds__(128) void attn_kernel(
    const _Float16* __restrict__ qb, const _Float16* __restrict__ kb,
    const _Float16* __restrict__ vb, float* __restrict__ out) {
  const int bh = blockIdx.y;  // b*4+h
  const int b = bh >> 2, h = bh & 3;
  const int nblk = blockIdx.x;  // 49 blocks of 64 rows
  const int tid = threadIdx.x;
  const int w = tid >> 5, l = tid & 31;
  const int lane16 = l & 15, hi = l >> 4;

  __shared__ _Float16 lq[64 * 64];       // 8 KB q tile [n][d]
  __shared__ _Float16 lk[32 * 64];       // 4 KB k tile [m][d]
  __shared__ _Float16 lvT[64 * 32];      // 4 KB v tile [d][m] (global is [d][M])
  __shared__ _Float16 lp[4][16 * 32];    // 4 KB per-wave P staging

  // ---- stage q (contiguous 8 KB)
  {
    const _Float16* src = qb + ((size_t)bh * NPIX + nblk * 64) * HDP;
    for (int i = tid; i < 512; i += 128) stage16(src + i * 8, &lq[i * 8]);
    stage_join();
  }
  __syncthreads();

  const _Float16* qrow = &lq[(w * 16 + lane16) * HDP];
  v16h aq0 = lda(qrow, hi);        // d 0..31
  v16h aq1 = lda(qrow + 32, hi);   // d 32..63 (zero-padded beyond 48)

  v8f o0 = {}, o1 = {}, o2 = {};
  float mrun[8], lrun[8];
#pragma unroll
  for (int e = 0; e < 8; ++e) { mrun[e] = -1e30f; lrun[e] = 0.f; }

  const float scale = 0.14433756729740643f;  // 48^-0.5
  const _Float16* kbase = kb + (size_t)bh * MPAD * HDP;
  const _Float16* vbase = vb + (size_t)bh * HDP * MPAD;  // transposed [d][M]

#pragma unroll 1
  for (int mt = 0; mt < MPAD / 32; ++mt) {
    // prefetch next k tile into the cache hierarchy
    if (mt + 1 < MPAD / 32) {
      __builtin_prefetch(kbase + (size_t)(mt + 1) * 32 * HDP, 0, 1);
      __builtin_prefetch(vbase + (size_t)(mt + 1) * 32, 0, 1);
    }
    // stage k tile (contiguous 4 KB) and v tile (64 strided rows of 64B)
    {
      const _Float16* ksrc = kbase + (size_t)mt * 32 * HDP;
      for (int i = tid; i < 256; i += 128) stage16(ksrc + i * 8, &lk[i * 8]);
      const _Float16* vsrc = vbase + mt * 32;
      for (int i = tid; i < 256; i += 128) {
        int d = i >> 2, c = i & 3;  // 4 x 8-half chunks per d-row
        stage16(vsrc + (size_t)d * MPAD + c * 8, &lvT[d * 32 + c * 8]);
      }
      stage_join();
    }
    __syncthreads();

    // ---- S = q * k^T : two 16-col subtiles, K = 64 (2 wmma each)
    v8f s0 = {}, s1 = {};
    {
      const _Float16* kc0 = &lk[lane16 * HDP];
      const _Float16* kc1 = &lk[(16 + lane16) * HDP];
      s0 = wmma32f16(aq0, ldb(kc0, hi), s0);
      s0 = wmma32f16(aq1, ldb(kc0 + 32, hi), s0);
      s1 = wmma32f16(aq0, ldb(kc1, hi), s1);
      s1 = wmma32f16(aq1, ldb(kc1 + 32, hi), s1);
    }
    int col0 = mt * 32 + lane16;
    int col1 = col0 + 16;
#pragma unroll
    for (int e = 0; e < 8; ++e) {
      s0[e] = (col0 < MPIX) ? s0[e] * scale : -1e30f;
      s1[e] = (col1 < MPIX) ? s1[e] * scale : -1e30f;
    }
    // ---- online softmax (rows live in lane halves -> shfl-only reductions)
    float rm[8];
#pragma unroll
    for (int e = 0; e < 8; ++e) rm[e] = fmaxf(s0[e], s1[e]);
#pragma unroll
    for (int off = 8; off >= 1; off >>= 1)
#pragma unroll
      for (int e = 0; e < 8; ++e) rm[e] = fmaxf(rm[e], __shfl_xor(rm[e], off, 32));
    float al[8];
#pragma unroll
    for (int e = 0; e < 8; ++e) {
      float mn = fmaxf(mrun[e], rm[e]);
      al[e] = __expf(mrun[e] - mn);
      mrun[e] = mn;
    }
#pragma unroll
    for (int e = 0; e < 8; ++e) {
      s0[e] = __expf(s0[e] - mrun[e]);
      s1[e] = __expf(s1[e] - mrun[e]);
    }
    float rs[8];
#pragma unroll
    for (int e = 0; e < 8; ++e) rs[e] = s0[e] + s1[e];
#pragma unroll
    for (int off = 8; off >= 1; off >>= 1)
#pragma unroll
      for (int e = 0; e < 8; ++e) rs[e] += __shfl_xor(rs[e], off, 32);
#pragma unroll
    for (int e = 0; e < 8; ++e) {
      lrun[e] = lrun[e] * al[e] + rs[e];
      o0[e] *= al[e];
      o1[e] *= al[e];
      o2[e] *= al[e];
    }
    // ---- stage P (f16) and reload as A-fragment
    {
      _Float16* P = lp[w];
#pragma unroll
      for (int e = 0; e < 8; ++e) {
        int row = e + hi * 8;
        P[row * 32 + lane16] = (_Float16)s0[e];
        P[row * 32 + 16 + lane16] = (_Float16)s1[e];
      }
      asm volatile("s_wait_dscnt 0x0" ::: "memory");
      v16h ap = lda(&P[lane16 * 32], hi);
      // ---- O += P * V (K = 32, 3 d-subtiles)
      o0 = wmma32f16(ap, ldb(&lvT[lane16 * 32], hi), o0);
      o1 = wmma32f16(ap, ldb(&lvT[(16 + lane16) * 32], hi), o1);
      o2 = wmma32f16(ap, ldb(&lvT[(32 + lane16) * 32], hi), o2);
    }
    __syncthreads();
  }

  // ---- finalize: O /= l, scatter to (B, N, C) f32
#pragma unroll
  for (int e = 0; e < 8; ++e) {
    float inv = 1.0f / lrun[e];
    int n = nblk * 64 + w * 16 + e + hi * 8;
    size_t base = ((size_t)b * NPIX + n) * C_ + h * HD;
    out[base + lane16 + 0] = o0[e] * inv;
    out[base + lane16 + 16] = o1[e] * inv;
    out[base + lane16 + 32] = o2[e] * inv;
  }
}

// ------------------------------------------------------------------- launcher

extern "C" void kernel_launch(void* const* d_in, const int* in_sizes, int n_in,
                              void* d_out, int out_size, void* d_ws,
                              size_t ws_size, hipStream_t stream) {
  const float* x    = (const float*)d_in[0];
  const float* dw_q = (const float*)d_in[3];
  const float* bqg  = (const float*)d_in[4];
  const float* bqb  = (const float*)d_in[5];
  const float* bqm  = (const float*)d_in[6];
  const float* bqv  = (const float*)d_in[7];
  const float* pw_q = (const float*)d_in[8];
  const float* dw_kv = (const float*)d_in[9];
  const float* bkg  = (const float*)d_in[10];
  const float* bkb  = (const float*)d_in[11];
  const float* bkm  = (const float*)d_in[12];
  const float* bkv  = (const float*)d_in[13];
  const float* pw_kv = (const float*)d_in[14];
  const float* Wq = (const float*)d_in[15];
  const float* bq = (const float*)d_in[16];
  const float* Wk = (const float*)d_in[17];
  const float* bk = (const float*)d_in[18];
  const float* Wv = (const float*)d_in[19];
  const float* bv = (const float*)d_in[20];
  float* out = (float*)d_out;

  char* ws = (char*)d_ws;
  size_t off = 0;
  auto alloc = [&](size_t bytes) -> void* {
    void* p = ws + off;
    off += (bytes + 255) & ~((size_t)255);
    return p;
  };
  float* bnq_s = (float*)alloc(C_ * 4);
  float* bnq_b = (float*)alloc(C_ * 4);
  float* bnk_s = (float*)alloc(C_ * 4);
  float* bnk_b = (float*)alloc(C_ * 4);
  _Float16* WqE = (_Float16*)alloc((size_t)C_ * C_ * 2);
  _Float16* WkE = (_Float16*)alloc((size_t)C_ * C_ * 2);
  _Float16* WvE = (_Float16*)alloc((size_t)C_ * C_ * 2);
  _Float16* zq  = (_Float16*)alloc((size_t)BATCH * NPIX * C_ * 2);
  _Float16* zkv = (_Float16*)alloc((size_t)BATCH * MPIX * C_ * 2);
  size_t qbytes = (size_t)BATCH * NHEADS * NPIX * HDP * 2;
  size_t kbytes = (size_t)BATCH * NHEADS * MPAD * HDP * 2;
  _Float16* qbuf = (_Float16*)alloc(qbytes);
  _Float16* kbuf = (_Float16*)alloc(kbytes);
  _Float16* vbuf = (_Float16*)alloc(kbytes);

  // zero d-pad (48..63) and m-pad (784..799) regions (f16 zero == all-zero bits)
  (void)hipMemsetAsync(qbuf, 0, qbytes, stream);
  (void)hipMemsetAsync(kbuf, 0, kbytes, stream);
  (void)hipMemsetAsync(vbuf, 0, kbytes, stream);

  fold_bn_kernel<<<1, C_, 0, stream>>>(bqg, bqb, bqm, bqv, bkg, bkb, bkm, bkv,
                                       bnq_s, bnq_b, bnk_s, bnk_b);
  fold_w_kernel<<<dim3(C_, 3), C_, 0, stream>>>(Wq, Wk, Wv, pw_q, pw_kv, WqE,
                                                WkE, WvE);

  {
    int total = BATCH * NPIX * (C_ / 4);
    dwconv_bn_kernel<<<(total + 255) / 256, 256, 0, stream>>>(
        x, dw_q, bnq_s, bnq_b, zq, 1, HH);
  }
  {
    int total = BATCH * MPIX * (C_ / 4);
    dwconv_bn_kernel<<<(total + 255) / 256, 256, 0, stream>>>(
        x, dw_kv, bnk_s, bnk_b, zkv, 2, HH / 2);
  }

  gemm_qkv_kernel<<<dim3(BATCH * NPIX / 64, 3), 128, 0, stream>>>(
      zq, WqE, bq, qbuf, NPIX, NPIX, 0);
  gemm_qkv_kernel<<<dim3(BATCH * MPIX / 64, 3), 128, 0, stream>>>(
      zkv, WkE, bk, kbuf, MPIX, MPAD, 0);
  gemm_qkv_kernel<<<dim3(BATCH * MPIX / 64, 3), 128, 0, stream>>>(
      zkv, WvE, bv, vbuf, MPIX, MPAD, 1);

  attn_kernel<<<dim3(NPIX / 64, BATCH * NHEADS), 128, 0, stream>>>(qbuf, kbuf,
                                                                   vbuf, out);
}
